// TrimodalMultiheadAttention_9663676416804
// MI455X (gfx1250) — compile-verified, matrix operand order checked by reading, not run on hardware
//
#include <hip/hip_runtime.h>
#include <hip/hip_bf16.h>

// ---------------------------------------------------------------------------
// TrimodalMultiheadAttention for MI455X (gfx1250), bf16 WMMA pipeline + TDM.
// T=128, B=8, D=512, H=8, HD=64.  Dominant cost: 17.2 GFLOP trilinear einsum,
// computed as per-(head, a) GEMMs K*diag(q_a) @ V^T with fused bias+mean/max.
// Bias tile (contiguous 64KB) is staged to LDS by the Tensor Data Mover,
// overlapped with the WMMA work, with TDM padding for bank-conflict-free reads.
// ---------------------------------------------------------------------------

typedef __attribute__((ext_vector_type(16))) __bf16 v16bf;
typedef __attribute__((ext_vector_type(4)))  __bf16 v4bf;
typedef __attribute__((ext_vector_type(8)))  float  v8f;
typedef __attribute__((ext_vector_type(4)))  unsigned int u32x4;
typedef __attribute__((ext_vector_type(8)))  int  i32x8;
typedef __attribute__((ext_vector_type(4)))  int  i32x4;

#define TT   128
#define BB   8
#define DD   512
#define HH   8
#define HDD  64
#define SCALING 0.125f          // 64^-0.5

#if defined(__gfx1250__) && __has_builtin(__builtin_amdgcn_tensor_load_to_lds)
#define HAVE_TDM 1
#else
#define HAVE_TDM 0
#endif

// LDS row stride (floats) for the TDM-padded bias tile: 128 data + 4 pad.
#define BSTRIDE 132

// ISA 7.12.2: 16-bit A/B matrix element e (0..15) of the v16bf fragment ->
// K index, given lane-half (lanes 0-15 -> half=0, lanes 16-31 -> half=1).
__device__ __forceinline__ int kidx(int e, int half) {
    int vg = e >> 1;
    return ((vg & 3) * 2) + (e & 1) + half * 8 + ((vg >> 2) * 16);
}

__device__ __forceinline__ v8f wmma_bf16(v16bf a, v16bf b, v8f c) {
    return __builtin_amdgcn_wmma_f32_16x16x32_bf16(
        false, a, false, b, (short)0, c, false, false);
}

// ---------------------------------------------------------------------------
// Kernel A: packed in-projection GEMM [1024x512] @ W[1536x512]^T, epilogue
// scatters into per-head layouts: qf (f32, scaled), kb/vb (bf16).
// grid = (8 m-tiles of 128 rows, 96 n-tiles of 16 cols), 256 threads (8 waves)
// ---------------------------------------------------------------------------
__global__ __launch_bounds__(256) void inproj_kernel(
    const float* __restrict__ query, const float* __restrict__ keyM,
    const float* __restrict__ valueM, const float* __restrict__ W,
    const float* __restrict__ biasP,
    float* __restrict__ qf, __bf16* __restrict__ kb, __bf16* __restrict__ vb)
{
    const int tid  = threadIdx.x;
    const int wv   = tid >> 5;
    const int lane = tid & 31;
    const int hl   = lane >> 4;
    const int ln   = lane & 15;
    const int m0   = blockIdx.x * 128;
    const int n0   = blockIdx.y * 16;          // 0..1535
    const int region = n0 >> 9;                // 0=q, 1=k, 2=v
    const float* A = (region == 0) ? query : ((region == 1) ? keyM : valueM);

    __shared__ __attribute__((aligned(16))) __bf16 Ab[128 * 32];
    __shared__ __attribute__((aligned(16))) __bf16 Wb[16 * 32];

    v8f acc = {};
    for (int k0 = 0; k0 < DD; k0 += 32) {
        __syncthreads();
        // vectorized f32->bf16 staging: float4 loads, b64 LDS stores
        for (int c4 = tid; c4 < (128 * 32) / 4; c4 += 256) {
            int r  = c4 >> 3;
            int cc = (c4 & 7) * 4;
            float4 f = *(const float4*)(A + (size_t)(m0 + r) * DD + k0 + cc);
            v4bf pk = { (__bf16)f.x, (__bf16)f.y, (__bf16)f.z, (__bf16)f.w };
            *(v4bf*)(Ab + r * 32 + cc) = pk;
        }
        for (int c4 = tid; c4 < (16 * 32) / 4; c4 += 256) {
            int r  = c4 >> 3;
            int cc = (c4 & 7) * 4;
            float4 f = *(const float4*)(W + (size_t)(n0 + r) * DD + k0 + cc);
            v4bf pk = { (__bf16)f.x, (__bf16)f.y, (__bf16)f.z, (__bf16)f.w };
            *(v4bf*)(Wb + r * 32 + cc) = pk;
        }
        __syncthreads();
        v16bf af, wf;
#pragma unroll
        for (int e = 0; e < 16; ++e) {
            int kk = kidx(e, hl);
            af[e] = Ab[(16 * wv + ln) * 32 + kk];  // A[m, k]
            wf[e] = Wb[ln * 32 + kk];              // B[k, n] = W[n, k]
        }
        acc = wmma_bf16(af, wf, acc);
    }

    const int head = (n0 >> 6) & 7;
#pragma unroll
    for (int j = 0; j < 8; ++j) {
        int mrow = m0 + 16 * wv + j + hl * 8;   // C layout: VGPR j -> M=j / j+8
        int n    = n0 + ln;                      // N = lane
        float v  = acc[j] + biasP[n];
        int t  = mrow >> 3;                      // row = t*B + bb, B=8
        int bb = mrow & 7;
        int i  = bb * HH + head;
        int dd = n & 63;
        size_t off = ((size_t)i * TT + t) * HDD + dd;
        if (region == 0)      qf[off] = v * SCALING;
        else if (region == 1) kb[off] = (__bf16)v;
        else                  vb[off] = (__bf16)v;
    }
}

// ---------------------------------------------------------------------------
// Kernel B: per (head i, query pos a): M = (K*diag(q_a)) @ V^T  (128x128),
// + bias tile, fused mean+max over c, softmax over b, attn row = w @ q.
// grid = (128 a, 64 i), 256 threads (8 waves); wave w owns b-rows [16w,16w+16)
// ---------------------------------------------------------------------------
__global__ __launch_bounds__(256) void score_attn_kernel(
    const float* __restrict__ qf, const __bf16* __restrict__ kb,
    const __bf16* __restrict__ vb, const float* __restrict__ biasC,
    float* __restrict__ attnW)
{
    const int a    = blockIdx.x;
    const int i    = blockIdx.y;
    const int tid  = threadIdx.x;
    const int wv   = tid >> 5;
    const int lane = tid & 31;
    const int hl   = lane >> 4;
    const int ln   = lane & 15;

    __shared__ __attribute__((aligned(16))) __bf16 KsL[TT * HDD]; // (K*q_a)[b][t]
    __shared__ __attribute__((aligned(16))) __bf16 VsL[TT * HDD]; // V[c][t]
    __shared__ float  qr[HDD];
    __shared__ float  fusedS[TT];
    __shared__ float  wS[TT];
    __shared__ float  red[8];

    // bias tile for this (batch, a): contiguous [128 b][128 c] f32 (64KB).
    const float* bp = biasC + (((size_t)(i >> 3) * TT + a) * TT) * TT;

#if HAVE_TDM
    __shared__ __attribute__((aligned(16))) float biasLds[TT * BSTRIDE];
    if (wv == 0) {
        // Tensor DMA: 2D tile 128x128 f32, contiguous in memory, padded in LDS
        // by 4 dwords every 128 dwords (row stride 132) -> conflict-free reads.
        unsigned lds  = (unsigned)(size_t)(void*)&biasLds[0];
        unsigned long long ga = (unsigned long long)(size_t)bp;
        u32x4 g0;
        g0[0] = 1u;                               // count=1 (valid descriptor)
        g0[1] = lds;                              // lds_addr
        g0[2] = (unsigned)(ga & 0xFFFFFFFFu);     // global_addr[31:0]
        g0[3] = (unsigned)((ga >> 32) & 0x1FFFFFFu) | (2u << 30); // type=2
        i32x8 g1;
        g1[0] = (int)((2u << 16)                  // data_size = 4 bytes
                    | (1u << 20)                  // pad_enable
                    | (6u << 22)                  // pad_interval = 128 dwords
                    | (3u << 25));                // pad_amount = 4 dwords
        g1[1] = (int)(128u << 16);                // tensor_dim0 = 128
        g1[2] = (int)(128u << 16);                // tensor_dim1 = 128
        g1[3] = (int)(128u << 16);                // tile_dim0 = 128
        g1[4] = (int)128u;                        // tile_dim1 = 128
        g1[5] = (int)128u;                        // tensor_dim0_stride = 128
        g1[6] = 0;
        g1[7] = 0;
        i32x4 z4 = {};
        i32x8 z8 = {};
        __builtin_amdgcn_tensor_load_to_lds(g0, g1, z4, z4, z8, 0);
    }
#else
    __builtin_prefetch(bp + (size_t)(16 * wv + lane) * TT, 0, 1);
#endif

    if (tid < HDD) qr[tid] = qf[((size_t)i * TT + a) * HDD + tid];
    __syncthreads();
    // vectorized staging: 8B loads of bf16 K/V; scale K by q_a in f32
    for (int c4 = tid; c4 < (TT * HDD) / 4; c4 += 256) {
        int idx = c4 * 4;
        int t = idx & 63;
        v4bf kv = *(const v4bf*)(kb + (size_t)i * (TT * HDD) + idx);
        v4bf vv = *(const v4bf*)(vb + (size_t)i * (TT * HDD) + idx);
        v4bf ks;
#pragma unroll
        for (int e = 0; e < 4; ++e) ks[e] = (__bf16)((float)kv[e] * qr[t + e]);
        *(v4bf*)(KsL + idx) = ks;
        *(v4bf*)(VsL + idx) = vv;
    }
    __syncthreads();

    // A fragments: rows 16w..16w+15 of K*diag(q_a), K-depth 64 = 2 frags
    v16bf a0, a1;
#pragma unroll
    for (int e = 0; e < 16; ++e) {
        int kk = kidx(e, hl);
        a0[e] = KsL[(16 * wv + ln) * HDD + kk];
        a1[e] = KsL[(16 * wv + ln) * HDD + kk + 32];
    }

    v8f acc[8];
#pragma unroll
    for (int ct = 0; ct < 8; ++ct) {
        v16bf b0, b1;
#pragma unroll
        for (int e = 0; e < 16; ++e) {
            int kk = kidx(e, hl);
            b0[e] = VsL[(ct * 16 + ln) * HDD + kk];       // B[k=t][n=c]=V[c][t]
            b1[e] = VsL[(ct * 16 + ln) * HDD + kk + 32];
        }
        v8f z = {};
        z = wmma_bf16(a0, b0, z);
        z = wmma_bf16(a1, b1, z);
        acc[ct] = z;
    }

#if HAVE_TDM
    if (wv == 0) __builtin_amdgcn_s_wait_tensorcnt(0);  // TDM done -> LDS valid
    __syncthreads();
#endif

    // bias add + fused reduction (mean + max over c) per b-row
#pragma unroll
    for (int j = 0; j < 8; ++j) {
        int brow = 16 * wv + j + hl * 8;
        float s = 0.f, mx = -3.4e38f;
#pragma unroll
        for (int ct = 0; ct < 8; ++ct) {
#if HAVE_TDM
            float v = acc[ct][j] + biasLds[brow * BSTRIDE + ct * 16 + ln];
#else
            float v = acc[ct][j] + bp[(size_t)brow * TT + ct * 16 + ln];
#endif
            s += v;
            mx = fmaxf(mx, v);
        }
#pragma unroll
        for (int off = 1; off < 16; off <<= 1) {
            s  += __shfl_xor(s, off, 32);
            mx  = fmaxf(mx, __shfl_xor(mx, off, 32));
        }
        if (ln == 0) fusedS[brow] = s * (1.0f / (float)TT) + mx;
    }
    __syncthreads();

    // softmax over b (128 values); all 256 threads keep uniform barriers
    bool act = tid < TT;
    float xv = act ? fusedS[tid] : -3.4e38f;
    float mr = xv;
    for (int off = 1; off < 32; off <<= 1) mr = fmaxf(mr, __shfl_xor(mr, off, 32));
    if (lane == 0) red[wv] = mr;
    __syncthreads();
    float gmax = red[0];
#pragma unroll
    for (int k = 1; k < 8; ++k) gmax = fmaxf(gmax, red[k]);
    float ex = act ? __expf(xv - gmax) : 0.f;
    float sr = ex;
    for (int off = 1; off < 32; off <<= 1) sr += __shfl_xor(sr, off, 32);
    __syncthreads();
    if (lane == 0) red[wv] = sr;
    __syncthreads();
    float tot = 0.f;
#pragma unroll
    for (int k = 0; k < 8; ++k) tot += red[k];
    if (act) wS[tid] = ex / tot;
    __syncthreads();

    // attn[a, :] = sum_b w[b] * q[i, b, :]  (f32 q, already scaled)
    if (tid < HDD) {
        const float* qp = qf + (size_t)i * (TT * HDD) + tid;
        float o = 0.f;
        for (int b = 0; b < TT; ++b) o += wS[b] * qp[(size_t)b * HDD];
        int bb = i >> 3, head = i & 7;
        attnW[((size_t)a * BB + bb) * DD + head * HDD + tid] = o;
    }
}

// ---------------------------------------------------------------------------
// Kernel C: out-projection [1024x512] @ out_w[512x512]^T + out_b -> d_out
// grid = (8, 32), 256 threads
// ---------------------------------------------------------------------------
__global__ __launch_bounds__(256) void outproj_kernel(
    const float* __restrict__ X, const float* __restrict__ W,
    const float* __restrict__ biasP, float* __restrict__ out)
{
    const int tid  = threadIdx.x;
    const int wv   = tid >> 5;
    const int lane = tid & 31;
    const int hl   = lane >> 4;
    const int ln   = lane & 15;
    const int m0   = blockIdx.x * 128;
    const int n0   = blockIdx.y * 16;

    __shared__ __attribute__((aligned(16))) __bf16 Ab[128 * 32];
    __shared__ __attribute__((aligned(16))) __bf16 Wb[16 * 32];

    v8f acc = {};
    for (int k0 = 0; k0 < DD; k0 += 32) {
        __syncthreads();
        for (int c4 = tid; c4 < (128 * 32) / 4; c4 += 256) {
            int r  = c4 >> 3;
            int cc = (c4 & 7) * 4;
            float4 f = *(const float4*)(X + (size_t)(m0 + r) * DD + k0 + cc);
            v4bf pk = { (__bf16)f.x, (__bf16)f.y, (__bf16)f.z, (__bf16)f.w };
            *(v4bf*)(Ab + r * 32 + cc) = pk;
        }
        for (int c4 = tid; c4 < (16 * 32) / 4; c4 += 256) {
            int r  = c4 >> 3;
            int cc = (c4 & 7) * 4;
            float4 f = *(const float4*)(W + (size_t)(n0 + r) * DD + k0 + cc);
            v4bf pk = { (__bf16)f.x, (__bf16)f.y, (__bf16)f.z, (__bf16)f.w };
            *(v4bf*)(Wb + r * 32 + cc) = pk;
        }
        __syncthreads();
        v16bf af, wf;
#pragma unroll
        for (int e = 0; e < 16; ++e) {
            int kk = kidx(e, hl);
            af[e] = Ab[(16 * wv + ln) * 32 + kk];
            wf[e] = Wb[ln * 32 + kk];
        }
        acc = wmma_bf16(af, wf, acc);
    }
#pragma unroll
    for (int j = 0; j < 8; ++j) {
        int mrow = m0 + 16 * wv + j + hl * 8;
        int n    = n0 + ln;
        out[(size_t)mrow * DD + n] = acc[j] + biasP[n];
    }
}

// ---------------------------------------------------------------------------
extern "C" void kernel_launch(void* const* d_in, const int* in_sizes, int n_in,
                              void* d_out, int out_size, void* d_ws, size_t ws_size,
                              hipStream_t stream) {
    (void)in_sizes; (void)n_in; (void)out_size; (void)ws_size;
    const float* query = (const float*)d_in[0];   // [T,B,512]
    const float* keyM  = (const float*)d_in[1];   // [T,B,512]
    const float* value = (const float*)d_in[2];   // [T,B,512]
    const float* corr  = (const float*)d_in[3];   // [B,T,T,T]
    const float* Wio   = (const float*)d_in[4];   // [1536,512]
    const float* bio   = (const float*)d_in[5];   // [1536]
    const float* outW  = (const float*)d_in[6];   // [512,512]
    const float* outB  = (const float*)d_in[7];   // [512]
    float* out = (float*)d_out;                   // [T,B,512]

    char* ws = (char*)d_ws;
    float*  qf    = (float*)(ws);                      // 64*128*64 f32 = 2 MB
    __bf16* kb    = (__bf16*)(ws + (size_t)(2 << 20)); // 1 MB
    __bf16* vb    = (__bf16*)(ws + (size_t)(3 << 20)); // 1 MB
    float*  attnW = (float*)(ws + (size_t)(4 << 20));  // 2 MB

    inproj_kernel<<<dim3(8, 96), 256, 0, stream>>>(query, keyM, value, Wio, bio,
                                                   qf, kb, vb);
    score_attn_kernel<<<dim3(128, 64), 256, 0, stream>>>(qf, kb, vb, corr, attnW);
    outproj_kernel<<<dim3(8, 32), 256, 0, stream>>>(attnW, outW, outB, out);
}
